// MultiLevelMultiHeadedAttentionRegister_64415919506138
// MI455X (gfx1250) — matrix operand order repaired and checked by reading, not compiled, not run on hardware
//
#include <hip/hip_runtime.h>
#include <cstdint>
#include <cstddef>

// ---- problem constants (from reference) ----
#define BB 16
#define CC 6
#define TT 512
#define SS 8
#define FF 512
#define HH 8
#define DK 64

// ---- vector types (trivial, union-safe) ----
typedef __attribute__((ext_vector_type(16))) __bf16 v16bf;
typedef __attribute__((ext_vector_type(8)))  float  v8f;
typedef __attribute__((ext_vector_type(4)))  unsigned int u32x4;
typedef __attribute__((ext_vector_type(4)))  float        f32x4;

union V16 {
    v16bf v;
    unsigned short s[16];
    u32x4 q[2];
};
union U8BF {
    unsigned short s[8];
    u32x4 q;
};

// Native bf16 convert (RNE in hardware) -- avoids 3-op integer emulation.
__device__ __forceinline__ unsigned short f2bf_u(float f) {
    __bf16 h = (__bf16)f;
    unsigned short u;
    __builtin_memcpy(&u, &h, 2);
    return u;
}
__device__ __forceinline__ float bf2f(unsigned short s) {
    return __uint_as_float(((unsigned int)s) << 16);
}
__device__ __forceinline__ v8f vzero8() {
    v8f z = {0.f, 0.f, 0.f, 0.f, 0.f, 0.f, 0.f, 0.f};
    return z;
}

// ---------------------------------------------------------------------------
// Vectorized f32 -> bf16 conversion: 8 elements / thread (2x b128 in, b128 out)
// ---------------------------------------------------------------------------
__global__ __launch_bounds__(256) void cvt_bf16x8_kernel(
    const float* __restrict__ src, unsigned short* __restrict__ dst, int n8) {
    int i = blockIdx.x * blockDim.x + threadIdx.x;
    if (i >= n8) return;
    f32x4 a = *((const f32x4*)src + 2 * (size_t)i);
    f32x4 b = *((const f32x4*)src + 2 * (size_t)i + 1);
    U8BF u;
#pragma unroll
    for (int j = 0; j < 4; ++j) {
        u.s[j]     = f2bf_u(a[j]);
        u.s[4 + j] = f2bf_u(b[j]);
    }
    *((u32x4*)dst + i) = u.q;
}

// ---------------------------------------------------------------------------
// Tiny K/V projection: out[row][n] = dot(in[row][:], W[n][:]) + bias[n]
// rows = B*S = 128 -> negligible; plain float4 dot.
// ---------------------------------------------------------------------------
__global__ void kvproj_kernel(const float* __restrict__ in,
                              const float* __restrict__ W,
                              const float* __restrict__ bias,
                              float* __restrict__ out, int rows) {
    int idx = blockIdx.x * blockDim.x + threadIdx.x;
    if (idx >= rows * FF) return;
    int n = idx & (FF - 1);
    int row = idx >> 9;
    const f32x4* a = (const f32x4*)(in + (size_t)row * FF);
    const f32x4* w = (const f32x4*)(W + (size_t)n * FF);
    float sum = bias[n];
#pragma unroll 8
    for (int i = 0; i < FF / 4; ++i) {
        f32x4 x = a[i], y = w[i];
        sum += x[0] * y[0] + x[1] * y[1] + x[2] * y[2] + x[3] * y[3];
    }
    out[idx] = sum;
}

// ---------------------------------------------------------------------------
// Q projection (WMMA bf16, A pre-converted to bf16) + fused mean over C.
//   grid.x = B*(T/16) row tiles, grid.y = 2 column groups of 256.
//   block = 256 = 8 waves; wave owns 16 rows x 32 cols (2 accumulators).
// ---------------------------------------------------------------------------
__global__ __launch_bounds__(256) void qproj_kernel(
    const unsigned short* __restrict__ qx,      // (B,C,T,F) bf16
    const unsigned short* __restrict__ wq,      // (F,F) bf16, row = out col n
    const float* __restrict__ bq,               // (F)
    unsigned short* __restrict__ qbf,           // (B,T,C,F) bf16
    float* __restrict__ q1)                     // (B,T,F) f32
{
    const int lane = threadIdx.x & 31;
    const int wave = threadIdx.x >> 5;
    const int lm = lane & 15;
    const int hi = lane >> 4;
    const int bt = blockIdx.x;
    const int b  = bt >> 5;                     // T/16 = 32 tiles per batch
    const int t0 = (bt & 31) << 4;
    const int col0 = blockIdx.y * 256 + wave * 32 + lm;
    const int col1 = col0 + 16;
    const float bias0 = bq[col0];
    const float bias1 = bq[col1];

    v8f msum0 = vzero8();
    v8f msum1 = vzero8();

    for (int c = 0; c < CC; ++c) {
        const unsigned short* arow =
            qx + (((size_t)b * CC + c) * TT + (t0 + lm)) * FF;
        v8f acc0 = vzero8();
        v8f acc1 = vzero8();
#pragma unroll
        for (int k0 = 0; k0 < FF; k0 += 32) {
            V16 av;
            av.q[0] = *(const u32x4*)(arow + k0 + hi * 16);
            av.q[1] = *(const u32x4*)(arow + k0 + hi * 16 + 8);
            const unsigned short* b0 = wq + (size_t)col0 * FF + k0 + hi * 16;
            const unsigned short* b1 = wq + (size_t)col1 * FF + k0 + hi * 16;
            V16 bv0, bv1;
            bv0.q[0] = *(const u32x4*)(b0);
            bv0.q[1] = *(const u32x4*)(b0 + 8);
            bv1.q[0] = *(const u32x4*)(b1);
            bv1.q[1] = *(const u32x4*)(b1 + 8);
            acc0 = __builtin_amdgcn_wmma_f32_16x16x32_bf16(
                false, av.v, false, bv0.v, (short)0, acc0, false, false);
            acc1 = __builtin_amdgcn_wmma_f32_16x16x32_bf16(
                false, av.v, false, bv1.v, (short)0, acc1, false, false);
        }
#pragma unroll
        for (int r = 0; r < 8; ++r) {
            const int trow = t0 + r + hi * 8;
            const size_t base = (((size_t)b * TT + trow) * CC + c) * FF;
            const float v0 = acc0[r] + bias0;
            const float v1 = acc1[r] + bias1;
            qbf[base + col0] = f2bf_u(v0);
            qbf[base + col1] = f2bf_u(v1);
            msum0[r] += v0;
            msum1[r] += v1;
        }
    }
#pragma unroll
    for (int r = 0; r < 8; ++r) {
        const int trow = t0 + r + hi * 8;
        const size_t base = ((size_t)b * TT + trow) * FF;
        q1[base + col0] = msum0[r] * (1.0f / 6.0f);
        q1[base + col1] = msum1[r] * (1.0f / 6.0f);
    }
}

// ---------------------------------------------------------------------------
// Attention stages 1+2: one wave per (b,t,h); lane owns d = lane, lane+32.
// ---------------------------------------------------------------------------
__global__ __launch_bounds__(256) void attn_kernel(
    const float* __restrict__ q1,               // (B,T,F)
    const unsigned short* __restrict__ qbf,     // (B,T,C,F) bf16
    const float* __restrict__ kk,               // (B,S,F)
    const float* __restrict__ vv,               // (B,S,F)
    const int* __restrict__ mask,               // (B,1,T,S)
    unsigned short* __restrict__ xbf)           // (B,T,F) bf16
{
    const int lane = threadIdx.x & 31;
    const int gw = blockIdx.x * 8 + (threadIdx.x >> 5);
    const int h = gw & 7;
    const int t = (gw >> 3) & (TT - 1);
    const int b = gw >> 12;
    const int d0 = lane, d1 = lane + 32;
    const size_t qbase = ((size_t)b * TT + t) * FF + h * DK;

    const float qa0 = q1[qbase + d0];
    const float qa1 = q1[qbase + d1];

    // stage 1 scores over S=8
    float sc[SS];
#pragma unroll
    for (int s = 0; s < SS; ++s) {
        const size_t kb = ((size_t)b * SS + s) * FF + h * DK;
        float p = qa0 * kk[kb + d0] + qa1 * kk[kb + d1];
#pragma unroll
        for (int off = 16; off > 0; off >>= 1) p += __shfl_xor(p, off, 32);
        sc[s] = p * 0.125f;                      // 1/sqrt(64)
    }
    const int* mrow = mask + ((size_t)b * TT + t) * SS;
#pragma unroll
    for (int s = 0; s < SS; ++s)
        if (mrow[s] == 0) sc[s] = -__builtin_inff();

    float mx = sc[0];
#pragma unroll
    for (int s = 1; s < SS; ++s) mx = fmaxf(mx, sc[s]);
    float at[SS], esum = 0.f;
#pragma unroll
    for (int s = 0; s < SS; ++s) {
        at[s] = (sc[s] == -__builtin_inff()) ? 0.f : __expf(sc[s] - mx);
        esum += at[s];
    }
    const float inv1 = 1.0f / esum;

    float p0 = 0.f, p1 = 0.f;
#pragma unroll
    for (int s = 0; s < SS; ++s) {
        const size_t vb = ((size_t)b * SS + s) * FF + h * DK;
        const float w = at[s] * inv1;
        p0 += w * vv[vb + d0];
        p1 += w * vv[vb + d1];
    }

    // stage 2 over C=6 channels
    float qc0[CC], qc1[CC], s2[CC];
#pragma unroll
    for (int c = 0; c < CC; ++c) {
        const size_t qc = (((size_t)b * TT + t) * CC + c) * FF + h * DK;
        qc0[c] = bf2f(qbf[qc + d0]);
        qc1[c] = bf2f(qbf[qc + d1]);
        float p = p0 * qc0[c] + p1 * qc1[c];
#pragma unroll
        for (int off = 16; off > 0; off >>= 1) p += __shfl_xor(p, off, 32);
        s2[c] = p * 0.125f;
    }
    float mx2 = s2[0];
#pragma unroll
    for (int c = 1; c < CC; ++c) mx2 = fmaxf(mx2, s2[c]);
    float e2[CC], esum2 = 0.f;
#pragma unroll
    for (int c = 0; c < CC; ++c) { e2[c] = __expf(s2[c] - mx2); esum2 += e2[c]; }
    const float inv2 = 1.0f / esum2;

    float x0 = 0.f, x1 = 0.f;
#pragma unroll
    for (int c = 0; c < CC; ++c) {
        const float w = e2[c] * inv2;
        x0 += w * qc0[c];
        x1 += w * qc1[c];
    }
    xbf[qbase + d0] = f2bf_u(x0);
    xbf[qbase + d1] = f2bf_u(x1);
}

// ---------------------------------------------------------------------------
// Output projection (WMMA bf16): out = x @ Wo.T + bo, rows = B*T = 8192.
//   grid.x = 512 row tiles, grid.y = 2; wave owns 16 rows x 32 cols.
// ---------------------------------------------------------------------------
__global__ __launch_bounds__(256) void oproj_kernel(
    const unsigned short* __restrict__ xbf,     // (B*T, F) bf16
    const unsigned short* __restrict__ wo,      // (F,F) bf16
    const float* __restrict__ bo,               // (F)
    float* __restrict__ out)                    // (B*T, F) f32
{
    const int lane = threadIdx.x & 31;
    const int wave = threadIdx.x >> 5;
    const int lm = lane & 15;
    const int hi = lane >> 4;
    const int r0 = blockIdx.x << 4;
    const int col0 = blockIdx.y * 256 + wave * 32 + lm;
    const int col1 = col0 + 16;

    const unsigned short* arow = xbf + (size_t)(r0 + lm) * FF;
    v8f acc0 = vzero8();
    v8f acc1 = vzero8();
#pragma unroll
    for (int k0 = 0; k0 < FF; k0 += 32) {
        V16 av;
        av.q[0] = *(const u32x4*)(arow + k0 + hi * 16);
        av.q[1] = *(const u32x4*)(arow + k0 + hi * 16 + 8);
        const unsigned short* b0 = wo + (size_t)col0 * FF + k0 + hi * 16;
        const unsigned short* b1 = wo + (size_t)col1 * FF + k0 + hi * 16;
        V16 bv0, bv1;
        bv0.q[0] = *(const u32x4*)(b0);
        bv0.q[1] = *(const u32x4*)(b0 + 8);
        bv1.q[0] = *(const u32x4*)(b1);
        bv1.q[1] = *(const u32x4*)(b1 + 8);
        acc0 = __builtin_amdgcn_wmma_f32_16x16x32_bf16(
            false, av.v, false, bv0.v, (short)0, acc0, false, false);
        acc1 = __builtin_amdgcn_wmma_f32_16x16x32_bf16(
            false, av.v, false, bv1.v, (short)0, acc1, false, false);
    }
    const float bias0 = bo[col0];
    const float bias1 = bo[col1];
#pragma unroll
    for (int r = 0; r < 8; ++r) {
        const size_t base = (size_t)(r0 + r + hi * 8) * FF;
        out[base + col0] = acc0[r] + bias0;
        out[base + col1] = acc1[r] + bias1;
    }
}

// ---------------------------------------------------------------------------
extern "C" void kernel_launch(void* const* d_in, const int* in_sizes, int n_in,
                              void* d_out, int out_size, void* d_ws, size_t ws_size,
                              hipStream_t stream) {
    const float* query = (const float*)d_in[0];
    const float* key   = (const float*)d_in[1];
    const float* value = (const float*)d_in[2];
    const int*   mask  = (const int*)d_in[3];
    const float* Wq = (const float*)d_in[4];
    const float* bq = (const float*)d_in[5];
    const float* Wk = (const float*)d_in[6];
    const float* bk = (const float*)d_in[7];
    const float* Wv = (const float*)d_in[8];
    const float* bv = (const float*)d_in[9];
    const float* Wo = (const float*)d_in[10];
    const float* bo = (const float*)d_in[11];
    float* out = (float*)d_out;

    // workspace layout (bytes)
    char* ws = (char*)d_ws;
    unsigned short* qxbf = (unsigned short*)(ws + 0);           // B*C*T*F bf16 = 50331648
    unsigned short* qbf  = (unsigned short*)(ws + 50331648);    // B*T*C*F bf16 = 50331648
    float*          q1   = (float*)(ws + 100663296);            // B*T*F f32    = 16777216
    float*          kf   = (float*)(ws + 117440512);            // B*S*F f32    = 262144
    float*          vf   = (float*)(ws + 117702656);            // B*S*F f32    = 262144
    unsigned short* xbf  = (unsigned short*)(ws + 117964800);   // B*T*F bf16   = 8388608
    unsigned short* wqbf = (unsigned short*)(ws + 126353408);   // F*F bf16     = 524288
    unsigned short* wobf = (unsigned short*)(ws + 126877696);   // F*F bf16     = 524288

    // one-shot conversions (memory-bound elementwise passes)
    const int q8 = (BB * CC * TT * FF) / 8;                     // 3145728
    const int w8 = (FF * FF) / 8;                               // 32768
    cvt_bf16x8_kernel<<<(q8 + 255) / 256, 256, 0, stream>>>(query, qxbf, q8);
    cvt_bf16x8_kernel<<<(w8 + 255) / 256, 256, 0, stream>>>(Wq, wqbf, w8);
    cvt_bf16x8_kernel<<<(w8 + 255) / 256, 256, 0, stream>>>(Wo, wobf, w8);

    kvproj_kernel<<<(BB * SS * FF) / 256, 256, 0, stream>>>(key,   Wk, bk, kf, BB * SS);
    kvproj_kernel<<<(BB * SS * FF) / 256, 256, 0, stream>>>(value, Wv, bv, vf, BB * SS);

    qproj_kernel<<<dim3(BB * (TT / 16), 2), 256, 0, stream>>>(qxbf, wqbf, bq, qbf, q1);

    attn_kernel<<<(BB * TT * HH) / 8, 256, 0, stream>>>(q1, qbf, kf, vf, mask, xbf);

    oproj_kernel<<<dim3((BB * TT) / 16, 2), 256, 0, stream>>>(xbf, wobf, bo, out);
}
